// MultiheadAttention_3401614098481
// MI455X (gfx1250) — matrix operand order repaired and checked by reading, not compiled, not run on hardware
//
#include <hip/hip_runtime.h>
#include <hip/hip_bf16.h>

#define EMBED 1024
#define NHEAD 16
#define HD    64
#define SEQ   2048
#define NB    2
#define MROWS (SEQ * NB) /* 4096 */
#define SBLK  64         /* keys per attention inner iteration */

// hd^-0.5 * log2(e): scores produced by Q@K^T land directly in the log2
// domain, so softmax exponentials are single v_exp_f32 (exp2) ops.
#define QSCALE (0.125f * 1.4426950408889634f)

typedef _Float16 half4v __attribute__((ext_vector_type(4)));
typedef _Float16 half8  __attribute__((ext_vector_type(8)));
typedef _Float16 v16h   __attribute__((ext_vector_type(16)));
typedef float    v8f    __attribute__((ext_vector_type(8)));
typedef unsigned int v4u __attribute__((ext_vector_type(4)));
typedef int      v4i    __attribute__((ext_vector_type(4)));
typedef int      v8i    __attribute__((ext_vector_type(8)));

#if defined(__AMDGCN__) && __has_builtin(__builtin_amdgcn_tensor_load_to_lds)
#define HAVE_TDM 1
#else
#define HAVE_TDM 0
#endif
#if defined(__AMDGCN__) && __has_builtin(__builtin_amdgcn_permlane16)
#define HAVE_PERMLANE 1
#else
#define HAVE_PERMLANE 0
#endif

// maxnum without fmaxf's library overhead (single v_max_num_f32 + canonicalize)
#define fmax2(a, b) __builtin_elementwise_max((a), (b))

__device__ __forceinline__ float fast_exp2(float x) {
#if defined(__AMDGCN__) && __has_builtin(__builtin_amdgcn_exp2f)
  return __builtin_amdgcn_exp2f(x);
#else
  return exp2f(x);
#endif
}

__device__ __forceinline__ float fast_rcp(float x) {
#if defined(__AMDGCN__) && __has_builtin(__builtin_amdgcn_rcpf)
  return __builtin_amdgcn_rcpf(x);
#else
  return 1.0f / x;
#endif
}

__device__ __forceinline__ v8f wmma16x16x32(v16h a, v16h b, v8f c) {
  return __builtin_amdgcn_wmma_f32_16x16x32_f16(false, a, false, b, (short)0, c,
                                                false, false);
}

__device__ __forceinline__ v8f vzero8() {
  v8f z = {0.f, 0.f, 0.f, 0.f, 0.f, 0.f, 0.f, 0.f};
  return z;
}

// ---- cross-lane helpers (16-lane-half row reductions, C-layout rows) -------
__device__ __forceinline__ float permx16(float v, unsigned s0, unsigned s1) {
#if HAVE_PERMLANE
  int i = __float_as_int(v);
  int r = __builtin_amdgcn_permlane16(i, i, (int)s0, (int)s1, false, false);
  return __int_as_float(r);
#else
  (void)s0; (void)s1;
  return v;
#endif
}

__device__ __forceinline__ float rowmax16(float v) {
#if HAVE_PERMLANE
  v = fmax2(v, permx16(v, 0x67452301u, 0xEFCDAB89u)); // xor 1
  v = fmax2(v, permx16(v, 0x54761032u, 0xDCFE98BAu)); // xor 2
  v = fmax2(v, permx16(v, 0x32107654u, 0xBA98FEDCu)); // xor 4
  v = fmax2(v, permx16(v, 0xFEDCBA98u, 0x76543210u)); // xor 8
#else
  v = fmax2(v, __shfl_xor(v, 1, 32));
  v = fmax2(v, __shfl_xor(v, 2, 32));
  v = fmax2(v, __shfl_xor(v, 4, 32));
  v = fmax2(v, __shfl_xor(v, 8, 32));
#endif
  return v;
}

__device__ __forceinline__ float rowbcast0(float v) {
#if HAVE_PERMLANE
  return permx16(v, 0x00000000u, 0x00000000u); // every lane reads row-lane 0
#else
  return __shfl(v, (int)(threadIdx.x & 16), 32);
#endif
}

// ---- WMMA fragment loaders (ISA 7.12.2 layouts) ----------------------------
__device__ __forceinline__ v16h load_a_frag(const _Float16* __restrict__ base,
                                            int lda, int row0, int k0) {
  const int lane = threadIdx.x & 31;
  const _Float16* p =
      base + (size_t)(row0 + (lane & 15)) * lda + k0 + ((lane >> 4) << 3);
  half8 lo = *(const half8*)(p);
  half8 hi = *(const half8*)(p + 16);
  return __builtin_shufflevector(lo, hi, 0, 1, 2, 3, 4, 5, 6, 7, 8, 9, 10, 11,
                                 12, 13, 14, 15);
}

// column n of B contiguous in `base` (row-major transpose of B, ld = ldb)
__device__ __forceinline__ v16h load_b_frag(const _Float16* base, int ldb,
                                            int col0, int k0) {
  const int lane = threadIdx.x & 31;
  const _Float16* p =
      base + (size_t)(col0 + (lane & 15)) * ldb + k0 + ((lane >> 4) << 3);
  half8 lo = *(const half8*)(p);
  half8 hi = *(const half8*)(p + 16);
  return __builtin_shufflevector(lo, hi, 0, 1, 2, 3, 4, 5, 6, 7, 8, 9, 10, 11,
                                 12, 13, 14, 15);
}

// ---- TDM: 2D tile load Global -> LDS (D# per ISA cdna5 ch.8) ---------------
#if HAVE_TDM
__device__ __forceinline__ void tdm_load_2d(unsigned lds_off, const void* gptr,
                                            unsigned tensor_d0, unsigned tensor_d1,
                                            unsigned tile_d0, unsigned tile_d1,
                                            unsigned stride0) {
  unsigned long long ga = (unsigned long long)gptr;
  v4u g0;
  g0[0] = 1u;                                           // count=1, user mode
  g0[1] = lds_off;                                      // lds_addr (bytes)
  g0[2] = (unsigned)(ga & 0xFFFFFFFFu);                 // global_addr[31:0]
  g0[3] = (unsigned)((ga >> 32) & 0x1FFFFFFu) | (2u << 30); // [56:32] | type=2
  v8i g1;
  g1[0] = (int)(1u << 16);                              // data_size=1 (2 bytes)
  g1[1] = (int)(tensor_d0 << 16);                       // tensor_dim0[15:0]
  g1[2] = (int)((tensor_d0 >> 16) | (tensor_d1 << 16)); // td0[31:16], td1[15:0]
  g1[3] = (int)((tensor_d1 >> 16) | (tile_d0 << 16));   // td1[31:16], tile0
  g1[4] = (int)(tile_d1 & 0xFFFFu);                     // tile1; tile2=0
  g1[5] = (int)stride0;                                 // dim0 stride[31:0]
  g1[6] = 0;                                            // stride0 hi, stride1 lo
  g1[7] = 0;
  v4i z4 = {0, 0, 0, 0};
#if __clang_major__ >= 23
  v8i z8 = {0, 0, 0, 0, 0, 0, 0, 0};
  __builtin_amdgcn_tensor_load_to_lds(g0, g1, z4, z4, z8, 0);
#else
  __builtin_amdgcn_tensor_load_to_lds(g0, g1, z4, z4, 0);
#endif
}
#endif

// ---------------------------------------------------------------- convert ---
__global__ void __launch_bounds__(256) cvt_f32_f16(const float* __restrict__ src,
                                                   _Float16* __restrict__ dst,
                                                   int n4) {
  int i = blockIdx.x * blockDim.x + threadIdx.x;
  const int stride = gridDim.x * blockDim.x;
  for (; i < n4; i += stride) {
    float4 v = ((const float4*)src)[i];
    half4v o = {(_Float16)v.x, (_Float16)v.y, (_Float16)v.z, (_Float16)v.w};
    ((half4v*)dst)[i] = o;
  }
}

// ---------------------------------------------------------------- GEMM ------
// C[row,col] = sum_k A[row,k]*W[col,k] + bias[col]; M=4096, N=1024, K=1024.
// MODE 0: Q16 [N,H,L,64] * (hd^-0.5 * log2e)   MODE 1: K16 [N,H,L,64]
// MODE 2: Vt [N,H,64,L]                        MODE 3: f32 out [M,E]
template <int MODE>
__global__ void __launch_bounds__(128) gemm16(const _Float16* __restrict__ A,
                                              const _Float16* __restrict__ W,
                                              const float* __restrict__ bias,
                                              void* __restrict__ dst) {
  const int w    = threadIdx.x >> 5;
  const int lane = threadIdx.x & 31;
  const int row0 = blockIdx.y * 128 + w * 32;
  const int col0 = blockIdx.x * 64;

  v8f acc[2][4];
#pragma unroll
  for (int i = 0; i < 2; ++i)
#pragma unroll
    for (int j = 0; j < 4; ++j) acc[i][j] = vzero8();

#pragma unroll 2
  for (int k0 = 0; k0 < EMBED; k0 += 32) {
    v16h a0 = load_a_frag(A, EMBED, row0, k0);
    v16h a1 = load_a_frag(A, EMBED, row0 + 16, k0);
#pragma unroll
    for (int j = 0; j < 4; ++j) {
      v16h b    = load_b_frag(W, EMBED, col0 + j * 16, k0);
      acc[0][j] = wmma16x16x32(a0, b, acc[0][j]);
      acc[1][j] = wmma16x16x32(a1, b, acc[1][j]);
    }
  }

  const int cbase = lane & 15;
  const int rbase = (lane >> 4) << 3;
#pragma unroll
  for (int i = 0; i < 2; ++i) {
#pragma unroll
    for (int j = 0; j < 4; ++j) {
      const int col = col0 + j * 16 + cbase;
      const float b = bias[col];
#pragma unroll
      for (int r = 0; r < 8; ++r) {
        const int row = row0 + i * 16 + rbase + r;
        float v       = acc[i][j][r] + b;
        if (MODE == 0 || MODE == 1) {
          if (MODE == 0) v *= QSCALE;
          const int l = row >> 1, n = row & 1;
          const int h = col >> 6, d = col & (HD - 1);
          ((_Float16*)dst)[(((size_t)(n * NHEAD + h) * SEQ + l) << 6) + d] =
              (_Float16)v;
        } else if (MODE == 2) {
          const int s = row >> 1, n = row & 1;
          const int h = col >> 6, d = col & (HD - 1);
          ((_Float16*)dst)[((size_t)(n * NHEAD + h) * HD + d) * SEQ + s] =
              (_Float16)v;
        } else {
          ((float*)dst)[(size_t)row * EMBED + col] = v;
        }
      }
    }
  }
}

// ------------------------------------------------------------- attention ----
// Flash attention per (n,h); block = 4 waves sharing TDM-staged 64x64 K/V
// tiles in double-buffered LDS; each wave owns 16 query rows and consumes 64
// keys per iteration (18 WMMAs). grid = (N*H, L/64). Scores arrive pre-scaled
// by log2e -> exp2-domain online softmax; row sums ride a ones-column WMMA.
__global__ void __launch_bounds__(128) attn_flash(const _Float16* __restrict__ Q,
                                                  const _Float16* __restrict__ K,
                                                  const _Float16* __restrict__ Vt,
                                                  _Float16* __restrict__ Ctx) {
  __shared__ _Float16 kbuf[2][SBLK * HD];   // K tile: [64 s][64 d]
  __shared__ _Float16 vbuf[2][HD * SBLK];   // V tile: [64 d][64 s]
  __shared__ _Float16 plds[4][16 * SBLK];   // per-wave P bounce [16 l][64 s]

  const int w      = threadIdx.x >> 5;
  const int lane   = threadIdx.x & 31;
  const int nh     = blockIdx.x;
  const int row_q0 = blockIdx.y * 64 + w * 16;

  const _Float16* Qp = Q + (size_t)nh * SEQ * HD;   // [L,64]
  const _Float16* Kp = K + (size_t)nh * SEQ * HD;   // [L,64]
  const _Float16* Vp = Vt + (size_t)nh * HD * SEQ;  // [64,L]

  const v16h aq0 = load_a_frag(Qp, HD, row_q0, 0);
  const v16h aq1 = load_a_frag(Qp, HD, row_q0, 32);

  const int cbase = lane & 15;
  const int rbase = (lane >> 4) << 3;

  // B-fragment of the implicit ones-column (col 0 of tile j==4).
  v16h bones;
  {
    const _Float16 one = (_Float16)((cbase == 0) ? 1.0f : 0.0f);
#pragma unroll
    for (int e = 0; e < 16; ++e) bones[e] = one;
  }

  float mrun[8];
#pragma unroll
  for (int r = 0; r < 8; ++r) mrun[r] = -1e30f;
  v8f o[5];  // o[0..3]: 16x64 output; o[4] col0: running row-sum of P
#pragma unroll
  for (int j = 0; j < 5; ++j) o[j] = vzero8();

  _Float16* myl = plds[w];
  const int NIT = SEQ / SBLK;  // 32

  auto stage = [&](int it) {
    const int s0  = it * SBLK;
    const int buf = it & 1;
#if HAVE_TDM
    if (w == 0) {
      tdm_load_2d((unsigned)(unsigned long long)(const void*)&kbuf[buf][0],
                  Kp + (size_t)s0 * HD, HD, SEQ, HD, SBLK, HD);
      tdm_load_2d((unsigned)(unsigned long long)(const void*)&vbuf[buf][0],
                  Vp + s0, SEQ, HD, SBLK, HD, SEQ);
    }
#else
    const int t = threadIdx.x;  // 128 threads; 64B each
    // K tile is one contiguous 8KB run in global
    *(v16h*)&kbuf[buf][t * 32]      = *(const v16h*)(Kp + (size_t)s0 * HD + t * 32);
    *(v16h*)&kbuf[buf][t * 32 + 16] = *(const v16h*)(Kp + (size_t)s0 * HD + t * 32 + 16);
    // V tile: row d strided by SEQ
    const int d = t >> 1, hh = (t & 1) * 32;
    *(v16h*)&vbuf[buf][d * SBLK + hh] =
        *(const v16h*)(Vp + (size_t)d * SEQ + s0 + hh);
    *(v16h*)&vbuf[buf][d * SBLK + hh + 16] =
        *(const v16h*)(Vp + (size_t)d * SEQ + s0 + hh + 16);
#endif
  };

  stage(0);
  for (int it = 0; it < NIT; ++it) {
#if HAVE_TDM
    if (w == 0) __builtin_amdgcn_s_wait_tensorcnt(0);
#endif
    __syncthreads();                      // tile `it` visible; prev readers done
    if (it + 1 < NIT) stage(it + 1);      // overlap DMA with compute
    const _Float16* kb = kbuf[it & 1];
    const _Float16* vb = vbuf[it & 1];

    // --- scores: 16(l) x 64(s), 8 WMMAs ---
    v8f sblk[4];
#pragma unroll
    for (int t = 0; t < 4; ++t) {
      sblk[t] = vzero8();
      v16h b0 = load_b_frag(kb, HD, t * 16, 0);
      v16h b1 = load_b_frag(kb, HD, t * 16, 32);
      sblk[t] = wmma16x16x32(aq0, b0, sblk[t]);
      sblk[t] = wmma16x16x32(aq1, b1, sblk[t]);
    }

    // --- exp2-domain online softmax (max only; sums ride the ones-WMMA) ---
    float alpha[8];
#pragma unroll
    for (int r = 0; r < 8; ++r) {
      const float bm = rowmax16(fmax2(fmax2(sblk[0][r], sblk[1][r]),
                                      fmax2(sblk[2][r], sblk[3][r])));
      const float mnew = fmax2(mrun[r], bm);
      alpha[r] = fast_exp2(mrun[r] - mnew);
      mrun[r]  = mnew;
#pragma unroll
      for (int t = 0; t < 4; ++t) sblk[t][r] = fast_exp2(sblk[t][r] - mnew);
    }

    // --- P (C-layout f32) -> LDS -> two A-fragments f16 ---
#pragma unroll
    for (int r = 0; r < 8; ++r)
#pragma unroll
      for (int t = 0; t < 4; ++t)
        myl[(rbase + r) * SBLK + t * 16 + cbase] = (_Float16)sblk[t][r];
    asm volatile("" ::: "memory");  // DS same-wave in-order; pin program order
    v16h pa0, pa1;
    {
      const _Float16* lp = myl + (lane & 15) * SBLK + ((lane >> 4) << 3);
      half8 lo0 = *(const half8*)lp;
      half8 hi0 = *(const half8*)(lp + 16);
      half8 lo1 = *(const half8*)(lp + 32);
      half8 hi1 = *(const half8*)(lp + 48);
      pa0 = __builtin_shufflevector(lo0, hi0, 0, 1, 2, 3, 4, 5, 6, 7, 8, 9, 10,
                                    11, 12, 13, 14, 15);
      pa1 = __builtin_shufflevector(lo1, hi1, 0, 1, 2, 3, 4, 5, 6, 7, 8, 9, 10,
                                    11, 12, 13, 14, 15);
    }
    asm volatile("" ::: "memory");

    // --- rescale and accumulate P@V (+ones), 10 WMMAs ---
#pragma unroll
    for (int j = 0; j < 5; ++j)
#pragma unroll
      for (int r = 0; r < 8; ++r) o[j][r] *= alpha[r];
#pragma unroll
    for (int j = 0; j < 4; ++j) {
      v16h bv0 = load_b_frag(vb, SBLK, j * 16, 0);
      v16h bv1 = load_b_frag(vb, SBLK, j * 16, 32);
      o[j] = wmma16x16x32(pa0, bv0, o[j]);
      o[j] = wmma16x16x32(pa1, bv1, o[j]);
    }
    o[4] = wmma16x16x32(pa0, bones, o[4]);
    o[4] = wmma16x16x32(pa1, bones, o[4]);
  }

  // --- normalize & write context [M = l*NB + n, E] f16 ---
  const int n = nh >> 4;
  const int h = nh & 15;
#pragma unroll
  for (int r = 0; r < 8; ++r) {
    const float inv = fast_rcp(rowbcast0(o[4][r]));
    const int l     = row_q0 + rbase + r;
    const size_t m  = (size_t)l * NB + n;
#pragma unroll
    for (int j = 0; j < 4; ++j) {
      const int e = h * HD + j * 16 + cbase;
      Ctx[m * EMBED + e] = (_Float16)(o[j][r] * inv);
    }
  }
}

// ---------------------------------------------------------------- launch ----
extern "C" void kernel_launch(void* const* d_in, const int* in_sizes, int n_in,
                              void* d_out, int out_size, void* d_ws,
                              size_t ws_size, hipStream_t stream) {
  const float* q   = (const float*)d_in[0];
  const float* k   = (const float*)d_in[1];
  const float* v   = (const float*)d_in[2];
  const float* ipw = (const float*)d_in[3];
  const float* ipb = (const float*)d_in[4];
  const float* opw = (const float*)d_in[5];
  const float* opb = (const float*)d_in[6];

  _Float16* Xq   = (_Float16*)d_ws;
  _Float16* Xk   = Xq + (size_t)MROWS * EMBED;
  _Float16* Xv   = Xk + (size_t)MROWS * EMBED;
  _Float16* Wh   = Xv + (size_t)MROWS * EMBED;      // [3E, E]
  _Float16* Woh  = Wh + (size_t)3 * EMBED * EMBED;  // [E, E]
  _Float16* Q16  = Woh + (size_t)EMBED * EMBED;     // [N,H,L,64]
  _Float16* K16  = Q16 + (size_t)MROWS * EMBED;
  _Float16* Vt16 = K16 + (size_t)MROWS * EMBED;     // [N,H,64,L]
  _Float16* Ctx  = Vt16 + (size_t)MROWS * EMBED;    // [M, E]

  auto cvt = [&](const float* s, _Float16* d, size_t n) {
    int n4     = (int)(n >> 2);
    int blocks = (n4 + 255) / 256;
    if (blocks > 2048) blocks = 2048;
    cvt_f32_f16<<<dim3(blocks), dim3(256), 0, stream>>>(s, d, n4);
  };
  cvt(q, Xq, (size_t)MROWS * EMBED);
  cvt(k, Xk, (size_t)MROWS * EMBED);
  cvt(v, Xv, (size_t)MROWS * EMBED);
  cvt(ipw, Wh, (size_t)3 * EMBED * EMBED);
  cvt(opw, Woh, (size_t)EMBED * EMBED);

  const dim3 gg(EMBED / 64, MROWS / 128);
  gemm16<0><<<gg, 128, 0, stream>>>(Xq, Wh, ipb, (void*)Q16);
  gemm16<1><<<gg, 128, 0, stream>>>(Xk, Wh + (size_t)EMBED * EMBED, ipb + EMBED,
                                    (void*)K16);
  gemm16<2><<<gg, 128, 0, stream>>>(Xv, Wh + (size_t)2 * EMBED * EMBED,
                                    ipb + 2 * EMBED, (void*)Vt16);

  attn_flash<<<dim3(NB * NHEAD, SEQ / 64), 128, 0, stream>>>(Q16, K16, Vt16,
                                                             Ctx);

  gemm16<3><<<gg, 128, 0, stream>>>(Ctx, Woh, opb, d_out);
}
